// MSARowAttentionWithPairBias_46918222742020
// MI455X (gfx1250) — matrix operand order repaired and checked by reading, not compiled, not run on hardware
//
#include <hip/hip_runtime.h>
#include <hip/hip_bf16.h>

// ---------------------------------------------------------------------------
// MSA row attention with pair bias, bf16 WMMA pipeline for gfx1250 (wave32).
// ---------------------------------------------------------------------------

typedef __attribute__((ext_vector_type(16))) __bf16 v16bf;
typedef __attribute__((ext_vector_type(8)))  float  v8f;

union BFrag { unsigned int u[8]; v16bf v; };

#define S_  128
#define N_  256
#define D_  256   // d_node
#define P_  128   // d_pair
#define H_  8
#define C_  32
#define HC_ 256
#define ROWS_ 32768  // S_*N_

__device__ __forceinline__ unsigned short f2bf(float x) {
  unsigned int u = __float_as_uint(x);
  u += 0x7fffu + ((u >> 16) & 1u);          // round-to-nearest-even
  return (unsigned short)(u >> 16);
}
__device__ __forceinline__ float bf2f(unsigned short s) {
  return __uint_as_float(((unsigned int)s) << 16);
}
__device__ __forceinline__ unsigned int pack2(float lo, float hi) {
  return (unsigned int)f2bf(lo) | ((unsigned int)f2bf(hi) << 16);
}

// ---------------------------------------------------------------------------
// 1) weights -> bf16  (Wbf = concat(qkv_w[768x256], gate_w[256x256]); W2bf = out_w)
// ---------------------------------------------------------------------------
__global__ void k_convw(const float* __restrict__ qkv_w, const float* __restrict__ gate_w,
                        const float* __restrict__ out_w,
                        unsigned short* __restrict__ Wbf, unsigned short* __restrict__ W2bf) {
  int i = blockIdx.x * 256 + threadIdx.x;
  if (i < 1024 * 256) {
    float v = (i < 768 * 256) ? qkv_w[i] : gate_w[i - 768 * 256];
    Wbf[i] = f2bf(v);
  } else {
    int j = i - 1024 * 256;
    W2bf[j] = f2bf(out_w[j]);
  }
}

// ---------------------------------------------------------------------------
// 2) LayerNorm(M_raw) -> bf16 rows.  One wave per 256-elem row.
// ---------------------------------------------------------------------------
__global__ __launch_bounds__(256) void k_ln_m(const float* __restrict__ M_raw,
                                              const float* __restrict__ g,
                                              const float* __restrict__ b,
                                              unsigned int* __restrict__ Mbf_u32) {
  int row  = (int)((blockIdx.x * 256 + threadIdx.x) >> 5);
  int lane = threadIdx.x & 31;
  const float* rp = M_raw + (size_t)row * D_;
  float4 a = *(const float4*)(rp + lane * 8);
  float4 c = *(const float4*)(rp + lane * 8 + 4);
  float x[8] = {a.x, a.y, a.z, a.w, c.x, c.y, c.z, c.w};
  float s = 0.f, sq = 0.f;
#pragma unroll
  for (int i = 0; i < 8; i++) { s += x[i]; sq += x[i] * x[i]; }
#pragma unroll
  for (int msk = 16; msk >= 1; msk >>= 1) {
    s  += __shfl_xor(s,  msk, 32);
    sq += __shfl_xor(sq, msk, 32);
  }
  float mu  = s * (1.0f / D_);
  float var = sq * (1.0f / D_) - mu * mu;
  float rs  = rsqrtf(var + 1e-5f);
  unsigned int* op = Mbf_u32 + (size_t)row * (D_ / 2) + lane * 4;
#pragma unroll
  for (int p = 0; p < 4; p++) {
    int d0 = lane * 8 + p * 2;
    float y0 = (x[p * 2 + 0] - mu) * rs * g[d0 + 0] + b[d0 + 0];
    float y1 = (x[p * 2 + 1] - mu) * rs * g[d0 + 1] + b[d0 + 1];
    op[p] = pack2(y0, y1);
  }
}

// ---------------------------------------------------------------------------
// 3) LayerNorm(Z) + bias[h,i,j] = sum_p Zn[i,j,p]*b_w[h,p].  One wave per (i,j).
// ---------------------------------------------------------------------------
__global__ __launch_bounds__(256) void k_ln_z_bias(const float* __restrict__ Z,
                                                   const float* __restrict__ g,
                                                   const float* __restrict__ b,
                                                   const float* __restrict__ bw,
                                                   float* __restrict__ bias) {
  int wid  = (int)((blockIdx.x * 256 + threadIdx.x) >> 5);   // 0..65535  (i*256+j)
  int lane = threadIdx.x & 31;
  const float* rp = Z + (size_t)wid * P_;
  float4 a = *(const float4*)(rp + lane * 4);
  float x[4] = {a.x, a.y, a.z, a.w};
  float s = 0.f, sq = 0.f;
#pragma unroll
  for (int i = 0; i < 4; i++) { s += x[i]; sq += x[i] * x[i]; }
#pragma unroll
  for (int msk = 16; msk >= 1; msk >>= 1) {
    s  += __shfl_xor(s,  msk, 32);
    sq += __shfl_xor(sq, msk, 32);
  }
  float mu  = s * (1.0f / P_);
  float var = sq * (1.0f / P_) - mu * mu;
  float rs  = rsqrtf(var + 1e-5f);
  float zn[4];
#pragma unroll
  for (int q = 0; q < 4; q++) {
    int d = lane * 4 + q;
    zn[q] = (x[q] - mu) * rs * g[d] + b[d];
  }
  int i = wid >> 8, j = wid & 255;
#pragma unroll
  for (int h = 0; h < H_; h++) {
    float p = 0.f;
#pragma unroll
    for (int q = 0; q < 4; q++) p += zn[q] * bw[h * P_ + lane * 4 + q];
#pragma unroll
    for (int msk = 16; msk >= 1; msk >>= 1) p += __shfl_xor(p, msk, 32);
    if (lane == 0) bias[h * 65536 + i * 256 + j] = p;
  }
}

// ---------------------------------------------------------------------------
// 4) WMMA NT-GEMM: [32768x256]bf16 x [1024x256]bf16^T -> Q(scaled)/K/V bf16 + gate bf16
//    per wave: 16x64 tile, K loop 8x32, 32 WMMAs.
// ---------------------------------------------------------------------------
__global__ __launch_bounds__(256) void k_gemm_qkvg(const unsigned short* __restrict__ Mbf,
                                                   const unsigned short* __restrict__ Wbf,
                                                   const float* __restrict__ gate_b,
                                                   unsigned short* __restrict__ Qs,
                                                   unsigned short* __restrict__ Kb,
                                                   unsigned short* __restrict__ Vb,
                                                   unsigned short* __restrict__ Gb) {
  int wid   = (int)((blockIdx.x * 256 + threadIdx.x) >> 5);
  int lane  = threadIdx.x & 31;
  int mtile = wid >> 4, ngroup = wid & 15;
  int m  = lane & 15;
  int hi = lane >> 4;            // 0 or 1
  int au = hi * 4, bu = hi * 8;  // u32 fragment offsets
  int arow = mtile * 16 + m;
  const unsigned int* Arow = (const unsigned int*)(Mbf + (size_t)arow * D_);
  const unsigned int* Brow[4];
#pragma unroll
  for (int t = 0; t < 4; t++) {
    int ncol = ngroup * 64 + t * 16 + m;
    Brow[t] = (const unsigned int*)(Wbf + (size_t)ncol * D_);
  }
  v8f acc[4] = {};
#pragma unroll
  for (int kb = 0; kb < 8; kb++) {
    BFrag A;
#pragma unroll
    for (int r = 0; r < 4; r++) {
      A.u[r]     = Arow[kb * 16 + bu + r];
      A.u[r + 4] = Arow[kb * 16 + bu + 4 + r];
    }
#pragma unroll
    for (int t = 0; t < 4; t++) {
      BFrag B;
#pragma unroll
      for (int r = 0; r < 8; r++) B.u[r] = Brow[t][kb * 16 + bu + r];
      acc[t] = __builtin_amdgcn_wmma_f32_16x16x32_bf16(false, A.v, false, B.v,
                                                       (short)0, acc[t], false, false);
    }
  }
  const float qscale = 0.17677669529663687f;  // 1/sqrt(32)
#pragma unroll
  for (int t = 0; t < 4; t++) {
    int e = ngroup * 64 + t * 16 + m;
#pragma unroll
    for (int r = 0; r < 8; r++) {
      int grow = mtile * 16 + r + hi * 8;
      int s = grow >> 8, n = grow & 255;
      float val = acc[t][r];
      if (e < 256) {
        int hq = e >> 5, cc = e & 31;
        Qs[(((size_t)(s * 8 + hq)) * 256 + n) * 32 + cc] = f2bf(val * qscale);
      } else if (e < 512) {
        int e2 = e - 256, hq = e2 >> 5, cc = e2 & 31;
        Kb[(((size_t)(s * 8 + hq)) * 256 + n) * 32 + cc] = f2bf(val);
      } else if (e < 768) {
        int e2 = e - 512, hq = e2 >> 5, cc = e2 & 31;
        Vb[(((size_t)(s * 8 + hq)) * 256 + n) * 32 + cc] = f2bf(val);
      } else {
        int eg = e - 768;
        float gv = 1.0f / (1.0f + __expf(-(val + gate_b[eg])));
        Gb[((size_t)(s * 256 + n)) * 256 + eg] = f2bf(gv);
      }
    }
  }
}

// ---------------------------------------------------------------------------
// 5) Attention: one block per (s,h). K->LDS [n][c], V->LDS transposed [c][n].
//    Per wave: 16-row tile, 16 QK^T WMMAs (bias as C), register softmax,
//    PV via per-wave LDS restage of weights, 16 more WMMAs. Gate in epilogue.
// ---------------------------------------------------------------------------
__global__ __launch_bounds__(256) void k_attn(const unsigned short* __restrict__ Qs,
                                              const unsigned short* __restrict__ Kb,
                                              const unsigned short* __restrict__ Vb,
                                              const float* __restrict__ bias,
                                              const unsigned short* __restrict__ Gb,
                                              unsigned short* __restrict__ Ob) {
  __shared__ unsigned short Ks[256 * 32];   // [n][c]
  __shared__ unsigned short Vt[32 * 256];   // [c][n]
  __shared__ unsigned short Wst[8 * 16 * 32];
  int s = blockIdx.x >> 3, h = blockIdx.x & 7;
  size_t base = ((size_t)(s * 8 + h)) * 256 * 32;
  int tid = threadIdx.x;
  {
    const unsigned int* kr = (const unsigned int*)(Kb + base + (size_t)tid * 32);
    unsigned int* kd = (unsigned int*)Ks + tid * 16;
#pragma unroll
    for (int r = 0; r < 16; r++) kd[r] = kr[r];
    const unsigned int* vr = (const unsigned int*)(Vb + base + (size_t)tid * 32);
    unsigned int vv[16];
#pragma unroll
    for (int r = 0; r < 16; r++) vv[r] = vr[r];
#pragma unroll
    for (int c2 = 0; c2 < 32; c2++)
      Vt[c2 * 256 + tid] = (unsigned short)(vv[c2 >> 1] >> ((c2 & 1) * 16));
  }
  __syncthreads();
  int wave = tid >> 5, lane = tid & 31;
  int m = lane & 15, hi = lane >> 4;
  int au = hi * 4, bu = hi * 8;

  for (int itl = 0; itl < 2; itl++) {
    int it = wave + itl * 8;                  // i-tile 0..15
    BFrag Aq;
    const unsigned int* qrow = (const unsigned int*)(Qs + base + (size_t)(it * 16 + m) * 32);
#pragma unroll
    for (int r = 0; r < 4; r++) { Aq.u[r] = qrow[bu + r]; Aq.u[r + 4] = qrow[bu + 4 + r]; }

    v8f acc[16];
#pragma unroll
    for (int jt = 0; jt < 16; jt++) {
      v8f c;
      const float* bp = bias + (size_t)h * 65536 + (size_t)(it * 16 + hi * 8) * 256 + jt * 16 + m;
#pragma unroll
      for (int r = 0; r < 8; r++) c[r] = bp[(size_t)r * 256];
      BFrag Bk;
      const unsigned int* kro = (const unsigned int*)Ks + (jt * 16 + m) * 16 + bu;
#pragma unroll
      for (int r = 0; r < 8; r++) Bk.u[r] = kro[r];
      acc[jt] = __builtin_amdgcn_wmma_f32_16x16x32_bf16(false, Aq.v, false, Bk.v,
                                                        (short)0, c, false, false);
    }

    // row softmax: row = r + 8*hi lives on the 16 lanes of this half-wave
    float rmax[8], rsum[8];
#pragma unroll
    for (int r = 0; r < 8; r++) rmax[r] = -3.0e38f;
#pragma unroll
    for (int jt = 0; jt < 16; jt++)
#pragma unroll
      for (int r = 0; r < 8; r++) rmax[r] = fmaxf(rmax[r], acc[jt][r]);
#pragma unroll
    for (int r = 0; r < 8; r++) {
#pragma unroll
      for (int msk = 1; msk < 16; msk <<= 1)
        rmax[r] = fmaxf(rmax[r], __shfl_xor(rmax[r], msk, 32));
      rsum[r] = 0.0f;
    }
#pragma unroll
    for (int jt = 0; jt < 16; jt++)
#pragma unroll
      for (int r = 0; r < 8; r++) {
        float e = __expf(acc[jt][r] - rmax[r]);
        acc[jt][r] = e;
        rsum[r] += e;
      }
#pragma unroll
    for (int r = 0; r < 8; r++) {
#pragma unroll
      for (int msk = 1; msk < 16; msk <<= 1) rsum[r] += __shfl_xor(rsum[r], msk, 32);
      rsum[r] = 1.0f / rsum[r];
    }

    // PV: restage softmax weights into A-fragment layout via per-wave LDS tile
    v8f oa = {}, obv = {};
    unsigned short* ws = Wst + wave * 512;    // [16][32]
#pragma unroll
    for (int jb = 0; jb < 8; jb++) {
#pragma unroll
      for (int p = 0; p < 2; p++) {
        int jt = jb * 2 + p;
#pragma unroll
        for (int r = 0; r < 8; r++)
          ws[(r + hi * 8) * 32 + p * 16 + m] = f2bf(acc[jt][r] * rsum[r]);
      }
      asm volatile("s_wait_dscnt 0" ::: "memory");   // same-wave cross-lane RAW
      BFrag Aw;
      const unsigned int* wr = (const unsigned int*)ws + m * 16;
#pragma unroll
      for (int r = 0; r < 4; r++) { Aw.u[r] = wr[bu + r]; Aw.u[r + 4] = wr[bu + 4 + r]; }
      BFrag Bv0, Bv1;
      const unsigned int* v0 = (const unsigned int*)Vt + m * 128 + jb * 16 + bu;
      const unsigned int* v1 = (const unsigned int*)Vt + (16 + m) * 128 + jb * 16 + bu;
#pragma unroll
      for (int r = 0; r < 8; r++) { Bv0.u[r] = v0[r]; Bv1.u[r] = v1[r]; }
      oa  = __builtin_amdgcn_wmma_f32_16x16x32_bf16(false, Aw.v, false, Bv0.v,
                                                    (short)0, oa, false, false);
      obv = __builtin_amdgcn_wmma_f32_16x16x32_bf16(false, Aw.v, false, Bv1.v,
                                                    (short)0, obv, false, false);
    }

    // epilogue: apply gate, store O bf16 at [s][n][h*32+cc]
#pragma unroll
    for (int r = 0; r < 8; r++) {
      int irow = it * 16 + r + hi * 8;
      size_t gbase = ((size_t)(s * 256 + irow)) * 256 + h * 32;
      float g0 = bf2f(Gb[gbase + m]);
      float g1 = bf2f(Gb[gbase + 16 + m]);
      Ob[gbase + m]      = f2bf(oa[r]  * g0);
      Ob[gbase + 16 + m] = f2bf(obv[r] * g1);
    }
  }
}

// ---------------------------------------------------------------------------
// 6) WMMA NT-GEMM out-proj: [32768x256]bf16 x out_w[256x256]bf16^T + bias + residual
// ---------------------------------------------------------------------------
__global__ __launch_bounds__(256) void k_gemm_out(const unsigned short* __restrict__ Obf,
                                                  const unsigned short* __restrict__ W2bf,
                                                  const float* __restrict__ out_bias,
                                                  const float* __restrict__ M_raw,
                                                  float* __restrict__ out) {
  int wid   = (int)((blockIdx.x * 256 + threadIdx.x) >> 5);
  int lane  = threadIdx.x & 31;
  int mtile = wid >> 2, ngroup = wid & 3;
  int m = lane & 15, hi = lane >> 4;
  int au = hi * 4, bu = hi * 8;
  int arow = mtile * 16 + m;
  const unsigned int* Arow = (const unsigned int*)(Obf + (size_t)arow * HC_);
  const unsigned int* Brow[4];
#pragma unroll
  for (int t = 0; t < 4; t++) {
    int ncol = ngroup * 64 + t * 16 + m;
    Brow[t] = (const unsigned int*)(W2bf + (size_t)ncol * HC_);
  }
  v8f acc[4] = {};
#pragma unroll
  for (int kb = 0; kb < 8; kb++) {
    BFrag A;
#pragma unroll
    for (int r = 0; r < 4; r++) {
      A.u[r]     = Arow[kb * 16 + bu + r];
      A.u[r + 4] = Arow[kb * 16 + bu + 4 + r];
    }
#pragma unroll
    for (int t = 0; t < 4; t++) {
      BFrag B;
#pragma unroll
      for (int r = 0; r < 8; r++) B.u[r] = Brow[t][kb * 16 + bu + r];
      acc[t] = __builtin_amdgcn_wmma_f32_16x16x32_bf16(false, A.v, false, B.v,
                                                       (short)0, acc[t], false, false);
    }
  }
#pragma unroll
  for (int t = 0; t < 4; t++) {
    int d = ngroup * 64 + t * 16 + m;
#pragma unroll
    for (int r = 0; r < 8; r++) {
      int grow = mtile * 16 + r + hi * 8;
      size_t idx = (size_t)grow * D_ + d;
      out[idx] = acc[t][r] + out_bias[d] + M_raw[idx];
    }
  }
}

// ---------------------------------------------------------------------------
// launch
// ---------------------------------------------------------------------------
extern "C" void kernel_launch(void* const* d_in, const int* in_sizes, int n_in,
                              void* d_out, int out_size, void* d_ws, size_t ws_size,
                              hipStream_t stream) {
  (void)in_sizes; (void)n_in; (void)out_size; (void)ws_size;
  const float* M_raw    = (const float*)d_in[0];
  const float* Z        = (const float*)d_in[1];
  const float* ln_m_g   = (const float*)d_in[2];
  const float* ln_m_b   = (const float*)d_in[3];
  const float* ln_z_g   = (const float*)d_in[4];
  const float* ln_z_b   = (const float*)d_in[5];
  const float* b_w      = (const float*)d_in[6];
  const float* qkv_w    = (const float*)d_in[7];
  const float* gate_w   = (const float*)d_in[8];
  const float* gate_b   = (const float*)d_in[9];
  const float* out_w    = (const float*)d_in[10];
  const float* out_bias = (const float*)d_in[11];
  float* out = (float*)d_out;

  char* ws = (char*)d_ws;
  size_t off = 0;
  auto alloc = [&](size_t bytes) -> void* {
    void* p = ws + off;
    off += (bytes + 255) & ~(size_t)255;
    return p;
  };
  unsigned short* Mbf  = (unsigned short*)alloc((size_t)ROWS_ * D_ * 2);       // 16.8 MB
  unsigned short* Wbf  = (unsigned short*)alloc((size_t)1024 * 256 * 2);       // 0.5 MB
  unsigned short* W2bf = (unsigned short*)alloc((size_t)256 * 256 * 2);        // 0.13 MB
  float*          bias = (float*)alloc((size_t)H_ * N_ * N_ * 4);              // 2.1 MB
  unsigned short* Qs   = (unsigned short*)alloc((size_t)S_ * H_ * N_ * C_ * 2);// 16.8 MB
  unsigned short* Kb   = (unsigned short*)alloc((size_t)S_ * H_ * N_ * C_ * 2);
  unsigned short* Vb   = (unsigned short*)alloc((size_t)S_ * H_ * N_ * C_ * 2);
  unsigned short* Gb   = (unsigned short*)alloc((size_t)ROWS_ * HC_ * 2);
  unsigned short* Ob   = (unsigned short*)alloc((size_t)ROWS_ * HC_ * 2);

  k_convw<<<1280, 256, 0, stream>>>(qkv_w, gate_w, out_w, Wbf, W2bf);
  k_ln_m<<<4096, 256, 0, stream>>>(M_raw, ln_m_g, ln_m_b, (unsigned int*)Mbf);
  k_ln_z_bias<<<8192, 256, 0, stream>>>(Z, ln_z_g, ln_z_b, b_w, bias);
  k_gemm_qkvg<<<4096, 256, 0, stream>>>(Mbf, Wbf, gate_b, Qs, Kb, Vb, Gb);
  k_attn<<<1024, 256, 0, stream>>>(Qs, Kb, Vb, bias, Gb, Ob);
  k_gemm_out<<<1024, 256, 0, stream>>>(Ob, W2bf, out_bias, M_raw, out);
}
